// MultiHeadCSGA_20753281974730
// MI455X (gfx1250) — compile-verified, matrix operand order checked by reading, not run on hardware
//
#include <hip/hip_runtime.h>
#include <hip/hip_bf16.h>
#include <math.h>

// ---------------------------------------------------------------------------
// MultiHead cross-scale group attention for MI455X (gfx1250), wave32 + WMMA
// + async global->LDS key staging.
// ---------------------------------------------------------------------------

typedef _Float16 h8   __attribute__((ext_vector_type(8)));
typedef _Float16 v16h __attribute__((ext_vector_type(16)));
typedef float    v8f  __attribute__((ext_vector_type(8)));

#define KSEG   5
#define CCH    256
#define HH     64
#define WW     64
#define NPIX   4096          // 64*64
#define NHD    8
#define HD     32
#define NKEY   2560          // K * 512
#define NKT    (NKEY / 16)   // 160 key tiles
#define SCALE_BASE 0.17677669529663687f   // 32^-0.5

__device__ __forceinline__ v16h make_frag(const _Float16* row, int half_id) {
    // 16-bit A/B fragment for one lane: two contiguous 16B chunks of a
    // row-major [*, 32] row: K = 8*half + {0..7} and 16 + 8*half + {0..7}.
    h8 lo = *(const h8*)(row + 8 * half_id);
    h8 hi = *(const h8*)(row + 16 + 8 * half_id);
    return __builtin_shufflevector(lo, hi,
           0, 1, 2, 3, 4, 5, 6, 7, 8, 9, 10, 11, 12, 13, 14, 15);
}

// ---------------------------------------------------------------------------
// 1) nearest-downsample delta [K,1,512,512] -> dmask [K,64,64]
// ---------------------------------------------------------------------------
__global__ void dmask_kernel(const float* __restrict__ delta, float* __restrict__ dmask) {
    int i = blockIdx.x * blockDim.x + threadIdx.x;
    if (i >= KSEG * NPIX) return;
    int k = i >> 12, p = i & 4095;
    int y = p >> 6, x = p & 63;
    dmask[i] = delta[(size_t)k * 512 * 512 + (size_t)(y * 8) * 512 + (size_t)(x * 8)];
}

// ---------------------------------------------------------------------------
// 2a) transpose+convert x [img][C][4096] f32 -> xt [img][4096][256] f16
//     (LDS 32x32 tile transpose, fully coalesced both directions)
// ---------------------------------------------------------------------------
__global__ __launch_bounds__(256) void xcvt_kernel(const float* __restrict__ x,
                                                   _Float16* __restrict__ xt) {
    int bid = blockIdx.x;            // img*1024 + ct*128 + pt
    int img = bid >> 10;
    int rem = bid & 1023;
    int ct = rem >> 7;               // 0..7   channel tile
    int pt = rem & 127;              // 0..127 pixel tile
    __shared__ _Float16 sh[32][33];

    const float* src = x + (size_t)img * CCH * NPIX + (size_t)(ct * 32) * NPIX + pt * 32;
    int tid = threadIdx.x;
#pragma unroll
    for (int i = 0; i < 4; ++i) {
        int idx = tid + i * 256;
        int cl = idx >> 5, pl = idx & 31;
        sh[cl][pl] = (_Float16)src[(size_t)cl * NPIX + pl];
    }
    __syncthreads();
    _Float16* dst = xt + (size_t)img * NPIX * CCH + (size_t)(pt * 32) * CCH + ct * 32;
#pragma unroll
    for (int i = 0; i < 4; ++i) {
        int idx = tid + i * 256;
        int pl = idx >> 5, cl = idx & 31;
        dst[(size_t)pl * CCH + cl] = sh[cl][pl];
    }
}

// 2b) convert Wq,Ws -> f16 [2][256][256] (already B^T layout: [o][c])
__global__ void wcvt_kernel(const float* __restrict__ Wq, const float* __restrict__ Ws,
                            _Float16* __restrict__ wf16) {
    int i = blockIdx.x * blockDim.x + threadIdx.x;
    if (i >= 2 * CCH * CCH) return;
    float v = (i < CCH * CCH) ? Wq[i] : Ws[i - CCH * CCH];
    wf16[i] = (_Float16)v;
}

// ---------------------------------------------------------------------------
// 3) projections via WMMA f32_16x16x32_f16:
//    out[pix][o] = sum_c xt[img][pix][c] * W[o][c] + b[o]
//    one wave per 16x16 output tile; per K-chunk: 4x b128 loads + 1 WMMA.
// ---------------------------------------------------------------------------
__global__ __launch_bounds__(32) void proj_kernel(
    const _Float16* __restrict__ xt, const _Float16* __restrict__ wf16,
    const float* __restrict__ bq, const float* __restrict__ bs,
    float* __restrict__ qproj, float* __restrict__ sproj)
{
    int bid   = blockIdx.x;          // img*4096 + ptile*16 + otile
    int img   = bid >> 12;
    int rem   = bid & 4095;
    int ptile = rem >> 4;
    int otile = rem & 15;
    int lane  = threadIdx.x;
    int half_id = lane >> 4;
    int mn      = lane & 15;

    const float* bias = (img == 0) ? bq : bs;
    float* outbuf = (img == 0) ? qproj
                               : (sproj + (size_t)(img - 1) * NPIX * CCH);

    const _Float16* arow = xt + (size_t)img * NPIX * CCH + (size_t)(ptile * 16 + mn) * CCH;
    const _Float16* wrow = wf16 + (size_t)((img == 0) ? 0 : CCH * CCH)
                                + (size_t)(otile * 16 + mn) * CCH;

    v8f acc = {};
#pragma unroll
    for (int kc = 0; kc < 8; ++kc) {
        int cb = kc * 32;
        v16h a = make_frag(arow + cb, half_id);
        v16h b = make_frag(wrow + cb, half_id);
        acc = __builtin_amdgcn_wmma_f32_16x16x32_f16(
            false, a, false, b, (short)0, acc, false, false);
    }

    float bval = bias[otile * 16 + mn];
#pragma unroll
    for (int r = 0; r < 8; ++r) {
        int m = r + 8 * half_id;
        outbuf[(size_t)(ptile * 16 + m) * CCH + otile * 16 + mn] = acc[r] + bval;
    }
}

// ---------------------------------------------------------------------------
// 4) l2-norm over 32 contiguous channels (one wave per (pixel,head) vector).
//    mode 0: q -> write f16 head-major qn[h][pix][32] scaled by scale*SCALE_BASE
//    mode 1: s -> normalize f32 in place
// ---------------------------------------------------------------------------
__global__ __launch_bounds__(128) void norm_kernel(
    float* __restrict__ proj, _Float16* __restrict__ qn,
    const float* __restrict__ scale_p, int mode, int nvec)
{
    int wave = blockIdx.x * 4 + (threadIdx.x >> 5);
    int d    = threadIdx.x & 31;
    if (wave >= nvec) return;

    float v = proj[(size_t)wave * 32 + d];
    float ss = v * v;
#pragma unroll
    for (int off = 1; off < 32; off <<= 1) ss += __shfl_xor(ss, off, 32);
    float denom = fmaxf(sqrtf(ss), 1e-12f);
    float nv = v / denom;

    if (mode == 0) {
        float sc = scale_p[0] * SCALE_BASE;
        int pix = wave >> 3, h = wave & 7;
        qn[(size_t)h * NPIX * HD + (size_t)pix * HD + d] = (_Float16)(nv * sc);
    } else {
        proj[(size_t)wave * 32 + d] = nv;
    }
}

// ---------------------------------------------------------------------------
// 5) patch centroids -> normalized keys (f16) + validity flags.
// ---------------------------------------------------------------------------
__global__ __launch_bounds__(32) void centroid_kernel(
    const float* __restrict__ snorm,   // [K][4096][256] (normalized)
    const float* __restrict__ dmask,   // [K][4096]
    _Float16* __restrict__ keys,       // [8][2560][32]
    float* __restrict__ maskflag)      // [2560] : 1 valid / 0 masked
{
    int bid = blockIdx.x;              // k*256 + s
    int k = bid >> 8, s = bid & 255;
    int prow = s >> 4, pcol = s & 15;
    int d = threadIdx.x;

    const float* dm = dmask + (size_t)k * NPIX;
    const float* sb = snorm + (size_t)k * NPIX * CCH;

    float w[16];
    int   pix[16];
    float cnt = 0.0f;
#pragma unroll
    for (int l = 0; l < 16; ++l) {
        int y = prow * 4 + (l >> 2), xx = pcol * 4 + (l & 3);
        pix[l] = y * 64 + xx;
        w[l]   = dm[pix[l]];
        cnt   += w[l];
    }
    if (d == 0) {
        maskflag[k * 512 + s]       = (cnt >= 1.0f) ? 1.0f : 0.0f;
        maskflag[k * 512 + 256 + s] = ((16.0f - cnt) >= 1.0f) ? 1.0f : 0.0f;
    }

    for (int h = 0; h < NHD; ++h) {
        float afg = 0.0f, abg = 0.0f;
#pragma unroll
        for (int l = 0; l < 16; ++l) {
            float v = sb[(size_t)pix[l] * CCH + h * HD + d];
            afg += w[l] * v;
            abg += (1.0f - w[l]) * v;
        }
        float sf = afg * afg, sg = abg * abg;
#pragma unroll
        for (int off = 1; off < 32; off <<= 1) {
            sf += __shfl_xor(sf, off, 32);
            sg += __shfl_xor(sg, off, 32);
        }
        float df = fmaxf(sqrtf(sf), 1e-12f);
        float db = fmaxf(sqrtf(sg), 1e-12f);
        size_t kb = (size_t)h * NKEY * HD + (size_t)(k * 512) * HD;
        keys[kb + (size_t)s * HD + d]         = (_Float16)(afg / df);
        keys[kb + (size_t)(256 + s) * HD + d] = (_Float16)(abg / db);
    }
}

// ---------------------------------------------------------------------------
// 6) attention: 8-wave blocks, 8 query tiles of one head share LDS-staged key
//    tiles (async global->LDS, double-buffered).  Logits via WMMA (K=32),
//    fused softmax foreground-mass; no max subtraction (|logit|<=0.18*|scale|).
// ---------------------------------------------------------------------------
__global__ __launch_bounds__(256) void attn_kernel(
    const _Float16* __restrict__ qn,     // [8][4096][32]
    const _Float16* __restrict__ keys,   // [8][2560][32]
    const float* __restrict__ maskflag,  // [2560]
    float* __restrict__ xo)              // [8][4096]
{
    __shared__ _Float16 kbuf[2][16 * HD];    // two 1KB key-tile buffers

    int bid  = blockIdx.x;                   // h*32 + mgroup
    int h    = bid >> 5;
    int mg   = bid & 31;
    int tid  = threadIdx.x;
    int wid  = tid >> 5;
    int lane = tid & 31;
    int half_id = lane >> 4;
    int mn      = lane & 15;
    int mt   = mg * 8 + wid;                 // this wave's 16-pixel tile

    const _Float16* qrow = qn + (size_t)h * NPIX * HD + (size_t)(mt * 16 + mn) * HD;
    v16h a = make_frag(qrow, half_id);

    const _Float16* kb = keys + (size_t)h * NKEY * HD;   // 64B per key
    unsigned long long kbase = (unsigned long long)(uintptr_t)kb;

    float den[8], num[8];
#pragma unroll
    for (int r = 0; r < 8; ++r) { den[r] = 0.0f; num[r] = 0.0f; }

    // prefetch key tile 0 into kbuf[0] (wave 0, async DMA to LDS)
    if (wid == 0) {
        unsigned lds0 = (unsigned)(uintptr_t)&kbuf[0][0] + (unsigned)(lane * 16);
        unsigned go   = (unsigned)(lane * 16);
        asm volatile("global_load_async_to_lds_b128 %0, %1, %2"
                     :: "v"(lds0), "v"(go), "s"(kbase) : "memory");
        asm volatile("global_load_async_to_lds_b128 %0, %1, %2"
                     :: "v"(lds0 + 512u), "v"(go + 512u), "s"(kbase) : "memory");
        asm volatile("s_wait_asynccnt 0" ::: "memory");
    }
    __syncthreads();

    for (int t = 0; t < NKT; ++t) {
        int cur = t & 1;
        // issue async copy of tile t+1 into the other buffer
        if (wid == 0 && (t + 1) < NKT) {
            unsigned ldsn = (unsigned)(uintptr_t)&kbuf[cur ^ 1][0] + (unsigned)(lane * 16);
            unsigned go   = (unsigned)((t + 1) * 1024 + lane * 16);
            asm volatile("global_load_async_to_lds_b128 %0, %1, %2"
                         :: "v"(ldsn), "v"(go), "s"(kbase) : "memory");
            asm volatile("global_load_async_to_lds_b128 %0, %1, %2"
                         :: "v"(ldsn + 512u), "v"(go + 512u), "s"(kbase) : "memory");
        }

        // compute on tile t from LDS
        const _Float16* krow = &kbuf[cur][0] + (size_t)mn * HD;
        v16h b = make_frag(krow, half_id);
        v8f c = {};
        v8f dmat = __builtin_amdgcn_wmma_f32_16x16x32_f16(
            false, a, false, b, (short)0, c, false, false);

        int key = t * 16 + mn;
        float valid = maskflag[key];
        float fg    = ((key & 511) < 256) ? 1.0f : 0.0f;
#pragma unroll
        for (int r = 0; r < 8; ++r) {
            float e = valid * __expf(dmat[r]);
            den[r] += e;
            num[r] += fg * e;
        }

        if (wid == 0) asm volatile("s_wait_asynccnt 0" ::: "memory");
        __syncthreads();
    }

#pragma unroll
    for (int r = 0; r < 8; ++r) {
        float dd = den[r], nn = num[r];
#pragma unroll
        for (int off = 1; off < 16; off <<= 1) {
            dd += __shfl_xor(dd, off, 32);
            nn += __shfl_xor(nn, off, 32);
        }
        if (mn == 0) {
            int m = r + 8 * half_id;
            xo[(size_t)h * NPIX + (size_t)(mt * 16 + m)] = nn / dd;
        }
    }
}

// ---------------------------------------------------------------------------
// 7) generic conv (NCHW, OIHW, stride 1, zero pad)
// ---------------------------------------------------------------------------
__global__ void conv_kernel(const float* __restrict__ in,
                            const float* __restrict__ w,
                            const float* __restrict__ bias,
                            float* __restrict__ out,
                            int IC, int OC, int KS, int pad)
{
    int i = blockIdx.x * blockDim.x + threadIdx.x;
    if (i >= OC * NPIX) return;
    int oc = i >> 12, p = i & 4095;
    int y = p >> 6, x = p & 63;

    float acc = bias[oc];
    const float* wr = w + (size_t)oc * IC * KS * KS;
    for (int ic = 0; ic < IC; ++ic) {
        const float* ip = in + (size_t)ic * NPIX;
        const float* wi = wr + (size_t)ic * KS * KS;
        for (int dy = 0; dy < KS; ++dy) {
            int yy = y + dy - pad;
            if (yy < 0 || yy >= HH) continue;
            for (int dx = 0; dx < KS; ++dx) {
                int xx = x + dx - pad;
                if (xx < 0 || xx >= WW) continue;
                acc += wi[dy * KS + dx] * ip[yy * 64 + xx];
            }
        }
    }
    out[i] = acc;
}

// ---------------------------------------------------------------------------
// 8) GroupNorm (groups=4): stats then normalize+ReLU in place
// ---------------------------------------------------------------------------
__global__ __launch_bounds__(256) void gn_stats_kernel(
    const float* __restrict__ in, float* __restrict__ stats, int C)
{
    int g = blockIdx.x;                     // 0..3
    int n = (C / 4) * NPIX;
    const float* base = in + (size_t)g * n;

    float s = 0.0f, s2 = 0.0f;
    for (int i = threadIdx.x; i < n; i += 256) {
        float v = base[i];
        s += v; s2 += v * v;
    }
#pragma unroll
    for (int off = 1; off < 32; off <<= 1) {
        s  += __shfl_xor(s,  off, 32);
        s2 += __shfl_xor(s2, off, 32);
    }
    __shared__ float sh0[8], sh1[8];
    int wid = threadIdx.x >> 5;
    if ((threadIdx.x & 31) == 0) { sh0[wid] = s; sh1[wid] = s2; }
    __syncthreads();
    if (threadIdx.x == 0) {
        float ts = 0.0f, ts2 = 0.0f;
        for (int wv = 0; wv < 8; ++wv) { ts += sh0[wv]; ts2 += sh1[wv]; }
        float mean = ts / (float)n;
        float var  = ts2 / (float)n - mean * mean;
        stats[g * 2]     = mean;
        stats[g * 2 + 1] = rsqrtf(var + 1e-5f);
    }
}

__global__ void gn_relu_kernel(float* __restrict__ buf,
                               const float* __restrict__ stats,
                               const float* __restrict__ gamma,
                               const float* __restrict__ beta, int C)
{
    int i = blockIdx.x * blockDim.x + threadIdx.x;
    if (i >= C * NPIX) return;
    int c = i >> 12;
    int g = c / (C / 4);
    float mean = stats[g * 2], inv = stats[g * 2 + 1];
    float v = (buf[i] - mean) * inv * gamma[c] + beta[c];
    buf[i] = fmaxf(v, 0.0f);
}

// ---------------------------------------------------------------------------
// host side
// ---------------------------------------------------------------------------
extern "C" void kernel_launch(void* const* d_in, const int* in_sizes, int n_in,
                              void* d_out, int out_size, void* d_ws, size_t ws_size,
                              hipStream_t stream) {
    (void)in_sizes; (void)n_in; (void)out_size;

    const float* x     = (const float*)d_in[0];
    const float* delta = (const float*)d_in[1];
    const float* Wq    = (const float*)d_in[2];
    const float* bq    = (const float*)d_in[3];
    const float* Ws    = (const float*)d_in[4];
    const float* bs    = (const float*)d_in[5];
    const float* scale = (const float*)d_in[6];
    const float* c1w   = (const float*)d_in[7];
    const float* c1b   = (const float*)d_in[8];
    const float* g1s   = (const float*)d_in[9];
    const float* g1b   = (const float*)d_in[10];
    const float* c2w   = (const float*)d_in[11];
    const float* c2b   = (const float*)d_in[12];
    const float* g2s   = (const float*)d_in[13];
    const float* g2b   = (const float*)d_in[14];
    const float* c3w   = (const float*)d_in[15];
    const float* c3b   = (const float*)d_in[16];
    const float* g3s   = (const float*)d_in[17];
    const float* g3b   = (const float*)d_in[18];
    float* out = (float*)d_out;

    char* ws = (char*)d_ws;
    size_t off = 0;
    auto alloc = [&](size_t bytes) -> char* {
        char* p = ws + off;
        off += (bytes + 255) & ~(size_t)255;
        return p;
    };
    _Float16*  xt       = (_Float16*) alloc((size_t)6 * NPIX * CCH * 2);      // 12.58 MB
    _Float16*  wf16     = (_Float16*) alloc((size_t)2 * CCH * CCH * 2);       //  0.26 MB
    float*     sproj    = (float*)    alloc((size_t)KSEG * NPIX * CCH * 4);   // 20.97 MB
    float*     qproj    = (float*)    alloc((size_t)NPIX * CCH * 4);          //  4.19 MB
    _Float16*  qn       = (_Float16*) alloc((size_t)NHD * NPIX * HD * 2);     //  2.10 MB
    _Float16*  keysb    = (_Float16*) alloc((size_t)NHD * NKEY * HD * 2);     //  1.31 MB
    float*     dmask    = (float*)    alloc((size_t)KSEG * NPIX * 4);
    float*     maskflag = (float*)    alloc((size_t)NKEY * 4);
    float*     xo       = (float*)    alloc((size_t)NHD * NPIX * 4);
    float*     c1out    = (float*)    alloc((size_t)16 * NPIX * 4);
    float*     c2out    = (float*)    alloc((size_t)64 * NPIX * 4);
    float*     stats    = (float*)    alloc(3 * 8 * 4);
    if (off > ws_size) return;   // workspace too small; nothing safe to do

    // 1) downsample mask
    dmask_kernel<<<(KSEG * NPIX + 255) / 256, 256, 0, stream>>>(delta, dmask);

    // 2) layout conversions for WMMA operands
    xcvt_kernel<<<6 * 1024, 256, 0, stream>>>(x, xt);
    wcvt_kernel<<<(2 * CCH * CCH + 255) / 256, 256, 0, stream>>>(Wq, Ws, wf16);

    // 3) projections (WMMA)
    proj_kernel<<<6 * 4096, 32, 0, stream>>>(xt, wf16, bq, bs, qproj, sproj);

    // 4) l2 norms
    {
        int nvq = NPIX * NHD;                // 32768 vectors
        norm_kernel<<<(nvq + 3) / 4, 128, 0, stream>>>(qproj, qn, scale, 0, nvq);
        int nvs = KSEG * NPIX * NHD;         // 163840 vectors
        norm_kernel<<<(nvs + 3) / 4, 128, 0, stream>>>(sproj, nullptr, scale, 1, nvs);
    }

    // 5) centroids -> keys + mask flags
    centroid_kernel<<<KSEG * 256, 32, 0, stream>>>(sproj, dmask, keysb, maskflag);

    // 6) attention (WMMA + async LDS key staging + fused softmax mass)
    attn_kernel<<<NHD * 32, 256, 0, stream>>>(qn, keysb, maskflag, xo);

    // 7) conv1 (8->16, 5x5) + GN + ReLU
    conv_kernel<<<(16 * NPIX + 255) / 256, 256, 0, stream>>>(xo, c1w, c1b, c1out, 8, 16, 5, 2);
    gn_stats_kernel<<<4, 256, 0, stream>>>(c1out, stats + 0, 16);
    gn_relu_kernel<<<(16 * NPIX + 255) / 256, 256, 0, stream>>>(c1out, stats + 0, g1s, g1b, 16);

    // 8) conv2 (16->64, 3x3) + GN + ReLU
    conv_kernel<<<(64 * NPIX + 255) / 256, 256, 0, stream>>>(c1out, c2w, c2b, c2out, 16, 64, 3, 1);
    gn_stats_kernel<<<4, 256, 0, stream>>>(c2out, stats + 8, 64);
    gn_relu_kernel<<<(64 * NPIX + 255) / 256, 256, 0, stream>>>(c2out, stats + 8, g2s, g2b, 64);

    // 9) conv3 (64->128, 3x3) + GN + ReLU -> d_out
    conv_kernel<<<(128 * NPIX + 255) / 256, 256, 0, stream>>>(c2out, c3w, c3b, out, 64, 128, 3, 1);
    gn_stats_kernel<<<4, 256, 0, stream>>>(out, stats + 16, 128);
    gn_relu_kernel<<<(128 * NPIX + 255) / 256, 256, 0, stream>>>(out, stats + 16, g3s, g3b, 128);
}
